// CIC_8521215115540
// MI455X (gfx1250) — matrix operand order repaired and checked by reading, not compiled, not run on hardware
//
#include <hip/hip_runtime.h>
#include <hip/hip_bf16.h>

// ---------------------------------------------------------------------------
// CurveAggregation (CIC) fused pipeline for MI455X (gfx1250, wave32, WMMA)
// B=16, C=256, N=8192, CN=128, CL=64, MID=128
// ---------------------------------------------------------------------------

typedef __attribute__((ext_vector_type(16))) __bf16 v16bf;
typedef __attribute__((ext_vector_type(8)))  __bf16 v8bf;
typedef __attribute__((ext_vector_type(8)))  float  v8f;

#define CIC_B   16
#define CIC_C   256
#define CIC_N   8192
#define CIC_CN  128
#define CIC_CL  64
#define CIC_MID 128
#define CIC_PT  32   // points per block in the main fused kernel

// ---------------- WMMA fragment helpers (CDNA5 layouts, wave32) ------------
// A matrix 16x32 bf16: lanes 0-15 rows M=0..15 (K groups 0-7,16-23),
// lanes 16-31 same rows (K groups 8-15,24-31). With K-contiguous LDS rows the
// 16 elements of a lane are two contiguous 16-byte chunks.
__device__ __forceinline__ v16bf load_A(const __bf16* base, int lane, int k0, int rs) {
  const __bf16* p = base + (size_t)(lane & 15) * rs + k0 + ((lane >> 4) << 3);
  v8bf lo = *(const v8bf*)p;
  v8bf hi = *(const v8bf*)(p + 16);
  v16bf a;
#pragma unroll
  for (int i = 0; i < 8; ++i) { a[i] = lo[i]; a[i + 8] = hi[i]; }
  return a;
}

// B matrix fragments are pre-packed in workspace: frag = 32 lanes x 16 bf16.
__device__ __forceinline__ v16bf load_Bg(const __bf16* base, int frag, int lane) {
  return *(const v16bf*)(base + (size_t)frag * 512 + lane * 16);
}

// C/D 16x16 f32: VGPR r holds M=r (lanes 0-15) and M=r+8 (lanes 16-31), N=lane%16.
__device__ __forceinline__ void store_D_bf16(__bf16* lds, int lane, v8f acc,
                                             int r0, int c0, int rs) {
  int half = lane >> 4;
  __bf16* p = lds + (size_t)(r0 + half * 8) * rs + c0 + (lane & 15);
#pragma unroll
  for (int rr = 0; rr < 8; ++rr) p[(size_t)rr * rs] = (__bf16)acc[rr];
}
__device__ __forceinline__ void store_D_f32(float* lds, int lane, v8f acc,
                                            int r0, int c0, int rs) {
  int half = lane >> 4;
  float* p = lds + (size_t)(r0 + half * 8) * rs + c0 + (lane & 15);
#pragma unroll
  for (int rr = 0; rr < 8; ++rr) p[(size_t)rr * rs] = acc[rr];
}

#define WMMA_BF16(A_, B_, ACC_) \
  (ACC_) = __builtin_amdgcn_wmma_f32_16x16x32_bf16(false, (A_), false, (B_), (short)0, (ACC_), false, false)

// ---------------- Prologue kernels ----------------------------------------

// att[b,n,l] = sum_c w_att[c] * curves[b,c,n,l]
__global__ __launch_bounds__(256) void cic_att_kernel(
    const float* __restrict__ curves, const float* __restrict__ watt,
    float* __restrict__ att) {
  int idx = blockIdx.x * 256 + threadIdx.x;          // < B*CN*CL = 131072
  int b = idx >> 13, r = idx & 8191;
  const float* cp = curves + ((size_t)b * CIC_C) * (CIC_CN * CIC_CL) + r;
  float s = 0.f;
#pragma unroll 4
  for (int c = 0; c < CIC_C; ++c) s += watt[c] * cp[(size_t)c * (CIC_CN * CIC_CL)];
  att[idx] = s;
}

// softmax over l for each (b,n) row of att
__global__ __launch_bounds__(256) void cic_softmax_l_kernel(
    const float* __restrict__ att, float* __restrict__ out, int rows) {
  int r = blockIdx.x * 256 + threadIdx.x;
  if (r >= rows) return;
  const float* p = att + (size_t)r * CIC_CL;
  float* o = out + (size_t)r * CIC_CL;
  float mx = -1e30f;
  for (int j = 0; j < CIC_CL; ++j) mx = fmaxf(mx, p[j]);
  float s = 0.f;
  for (int j = 0; j < CIC_CL; ++j) s += __expf(p[j] - mx);
  float inv = 1.f / s;
  for (int j = 0; j < CIC_CL; ++j) o[j] = __expf(p[j] - mx) * inv;
}

// softmax over n for each (b,l) column of att
__global__ __launch_bounds__(256) void cic_softmax_n_kernel(
    const float* __restrict__ att, float* __restrict__ out, int cols) {
  int idx = blockIdx.x * 256 + threadIdx.x;
  if (idx >= cols) return;
  int b = idx >> 6, l = idx & 63;
  const float* p = att + (size_t)b * (CIC_CN * CIC_CL) + l;
  float* o = out + (size_t)b * (CIC_CN * CIC_CL) + l;
  float mx = -1e30f;
  for (int n = 0; n < CIC_CN; ++n) mx = fmaxf(mx, p[n * CIC_CL]);
  float s = 0.f;
  for (int n = 0; n < CIC_CN; ++n) s += __expf(p[n * CIC_CL] - mx);
  float inv = 1.f / s;
  for (int n = 0; n < CIC_CN; ++n) o[n * CIC_CL] = __expf(p[n * CIC_CL] - mx) * inv;
}

// ci[b,c,n] = sum_l curves[b,c,n,l]*sml[b,n,l]; ct[b,c,l] = sum_n curves*smn
__global__ __launch_bounds__(128) void cic_cict_kernel(
    const float* __restrict__ curves, const float* __restrict__ sml,
    const float* __restrict__ smn, float* __restrict__ ci, float* __restrict__ ct) {
  __shared__ float tile[CIC_CN * CIC_CL];            // 32 KB
  int bc = blockIdx.x;                               // b*C + c
  int b = bc >> 8;
  const float* src = curves + (size_t)bc * (CIC_CN * CIC_CL);
  for (int i = threadIdx.x; i < CIC_CN * CIC_CL; i += 128) tile[i] = src[i];
  __syncthreads();
  {
    int n = threadIdx.x;
    const float* sl = sml + (size_t)b * (CIC_CN * CIC_CL) + n * CIC_CL;
    float s = 0.f;
#pragma unroll 4
    for (int l = 0; l < CIC_CL; ++l) s += tile[n * CIC_CL + l] * sl[l];
    ci[(size_t)bc * CIC_CN + n] = s;
  }
  if (threadIdx.x < CIC_CL) {
    int l = threadIdx.x;
    const float* sn = smn + (size_t)b * (CIC_CN * CIC_CL) + l;
    float s = 0.f;
#pragma unroll 4
    for (int n = 0; n < CIC_CN; ++n) s += tile[n * CIC_CL + l] * sn[n * CIC_CL];
    ct[(size_t)bc * CIC_CL + l] = s;
  }
}

// Y[b,m,n] = sum_k W[m,k] * X[b,k,n]   (tiny GEMMs: a, b, an, bl)
__global__ __launch_bounds__(256) void cic_mm_small_kernel(
    const float* __restrict__ W, const float* __restrict__ X,
    float* __restrict__ Y, int M, int K, int Nc, int total) {
  int idx = blockIdx.x * 256 + threadIdx.x;
  if (idx >= total) return;
  int b = idx / (M * Nc), r = idx % (M * Nc);
  int m = r / Nc, n = r % Nc;
  const float* wr = W + (size_t)m * K;
  const float* xc = X + (size_t)b * K * Nc + n;
  float s = 0.f;
#pragma unroll 4
  for (int k = 0; k < K; ++k) s += wr[k] * xc[(size_t)k * Nc];
  Y[idx] = s;
}

// Pack B[k,n] (given strides into an f32 source) into WMMA B fragments:
// frag f=(kt,nt): lane holds B(kt*32 + (j<8 ? (lane/16)*8 + j : 16 + (lane/16)*8 + j-8), nt*16 + lane%16), j=0..15.
__global__ __launch_bounds__(256) void cic_pack_b_kernel(
    const float* __restrict__ src, __bf16* __restrict__ dst,
    int K, int Nn, int sk, int sn, long bstride, int fpb, int total_frags) {
  int t = blockIdx.x * 256 + threadIdx.x;
  int fid = t >> 5;
  if (fid >= total_frags) return;
  int lane = t & 31;
  int b = fid / fpb, f = fid % fpb;
  int ntn = Nn >> 4;
  int kt = f / ntn, nt = f % ntn;
  int half = lane >> 4, col = nt * 16 + (lane & 15);
  const float* s = src + (long)b * bstride;
  __bf16* d = dst + (size_t)fid * 512 + lane * 16;
#pragma unroll
  for (int j = 0; j < 16; ++j) {
    int k = kt * 32 + (j < 8 ? half * 8 + j : 16 + half * 8 + (j - 8));
    d[j] = (__bf16)s[(long)k * sk + (long)col * sn];
  }
}

// ---------------- Main fused kernel ---------------------------------------
// Per block: b = blockIdx.y, 32 points. Pipeline:
//  XL = x^T wc^T -> logits_n=XL a (softmax) -> logits_l=XL b (softmax)
//  xi = smn an, xt = sml bl -> feat -> y = feat wd^T -> BN + residual + leaky
__global__ __launch_bounds__(256) void cic_main_kernel(
    const float* __restrict__ x,
    const __bf16* __restrict__ Bwc, const __bf16* __restrict__ Ba,
    const __bf16* __restrict__ Bb,  const __bf16* __restrict__ Ban,
    const __bf16* __restrict__ Bbl, const __bf16* __restrict__ Bwd,
    const float* __restrict__ gamma, const float* __restrict__ beta,
    const float* __restrict__ mean,  const float* __restrict__ var,
    float* __restrict__ out) {
  __shared__ __align__(16) char smem[55296];
  __bf16* xt   = (__bf16*)(smem);           // [32][264] x tile (p-major, pad 8)
  __bf16* feat = (__bf16*)(smem);           // alias: [32][264] (256 + pad)
  __bf16* xl   = (__bf16*)(smem + 16896);   // [32][136] x_logits bf16
  float*  lg   = (float* )(smem + 25600);   // [32][128] raw logits f32
  __bf16* smn  = (__bf16*)(smem + 41984);   // [32][136] softmax over n
  __bf16* sml  = (__bf16*)(smem + 50688);   // [32][72]  softmax over l

  const int b   = blockIdx.y;
  const int p0  = blockIdx.x * CIC_PT;
  const int tid = threadIdx.x;
  const int w = tid >> 5, lane = tid & 31;

  // ---- stage 1: load x tile (bf16, [p][c], rs=264) ----
  for (int i = tid; i < CIC_C * CIC_PT; i += 256) {
    int c = i >> 5, p = i & 31;
    xt[p * 264 + c] = (__bf16)x[((size_t)b * CIC_C + c) * CIC_N + p0 + p];
  }
  __syncthreads();

  // ---- XL[32p x 128m], K=256 ----
  for (int t = w; t < 16; t += 8) {
    int rt = t >> 3, mt = t & 7;
    v8f acc = {};
    const __bf16* ab = xt + rt * 16 * 264;
#pragma unroll
    for (int kc = 0; kc < 8; ++kc) {
      v16bf A = load_A(ab, lane, kc * 32, 264);
      v16bf Bf = load_Bg(Bwc, kc * 8 + mt, lane);
      WMMA_BF16(A, Bf, acc);
    }
    store_D_bf16(xl, lane, acc, rt * 16, mt * 16, 136);
  }
  __syncthreads();

  // ---- logits_n = XL @ a[b]  [32 x 128], K=128 ----
  const __bf16* Bab = Ba + (size_t)b * 32 * 512;
  for (int t = w; t < 16; t += 8) {
    int rt = t >> 3, nt = t & 7;
    v8f acc = {};
    const __bf16* ab = xl + rt * 16 * 136;
#pragma unroll
    for (int kc = 0; kc < 4; ++kc) {
      v16bf A = load_A(ab, lane, kc * 32, 136);
      v16bf Bf = load_Bg(Bab, kc * 8 + nt, lane);
      WMMA_BF16(A, Bf, acc);
    }
    store_D_f32(lg, lane, acc, rt * 16, nt * 16, 128);
  }
  __syncthreads();
  if (tid < CIC_PT) {
    float* row = lg + tid * 128;
    float mx = -1e30f;
    for (int j = 0; j < 128; ++j) mx = fmaxf(mx, row[j]);
    float s = 0.f;
    for (int j = 0; j < 128; ++j) { float e = __expf(row[j] - mx); row[j] = e; s += e; }
    float inv = 1.f / s;
    for (int j = 0; j < 128; ++j) smn[tid * 136 + j] = (__bf16)(row[j] * inv);
  }
  __syncthreads();

  // ---- logits_l = XL @ bmat[b]  [32 x 64], K=128 (one tile per wave) ----
  const __bf16* Bbb = Bb + (size_t)b * 16 * 512;
  {
    int rt = w >> 2, lt = w & 3;
    v8f acc = {};
    const __bf16* ab = xl + rt * 16 * 136;
#pragma unroll
    for (int kc = 0; kc < 4; ++kc) {
      v16bf A = load_A(ab, lane, kc * 32, 136);
      v16bf Bf = load_Bg(Bbb, kc * 4 + lt, lane);
      WMMA_BF16(A, Bf, acc);
    }
    store_D_f32(lg, lane, acc, rt * 16, lt * 16, 64);
  }
  __syncthreads();
  if (tid < CIC_PT) {
    float* row = lg + tid * 64;
    float mx = -1e30f;
    for (int j = 0; j < 64; ++j) mx = fmaxf(mx, row[j]);
    float s = 0.f;
    for (int j = 0; j < 64; ++j) { float e = __expf(row[j] - mx); row[j] = e; s += e; }
    float inv = 1.f / s;
    for (int j = 0; j < 64; ++j) sml[tid * 72 + j] = (__bf16)(row[j] * inv);
  }
  __syncthreads();

  // ---- xi = smn @ an[b] -> feat[:, 0:128], K=128 ----
  const __bf16* Banb = Ban + (size_t)b * 32 * 512;
  for (int t = w; t < 16; t += 8) {
    int rt = t >> 3, mt = t & 7;
    v8f acc = {};
    const __bf16* ab = smn + rt * 16 * 136;
#pragma unroll
    for (int kc = 0; kc < 4; ++kc) {
      v16bf A = load_A(ab, lane, kc * 32, 136);
      v16bf Bf = load_Bg(Banb, kc * 8 + mt, lane);
      WMMA_BF16(A, Bf, acc);
    }
    store_D_bf16(feat, lane, acc, rt * 16, mt * 16, 264);
  }
  // ---- xt = sml @ bl[b] -> feat[:, 128:256], K=64 ----
  const __bf16* Bblb = Bbl + (size_t)b * 16 * 512;
  for (int t = w; t < 16; t += 8) {
    int rt = t >> 3, mt = t & 7;
    v8f acc = {};
    const __bf16* ab = sml + rt * 16 * 72;
#pragma unroll
    for (int kc = 0; kc < 2; ++kc) {
      v16bf A = load_A(ab, lane, kc * 32, 72);
      v16bf Bf = load_Bg(Bblb, kc * 8 + mt, lane);
      WMMA_BF16(A, Bf, acc);
    }
    store_D_bf16(feat, lane, acc, rt * 16, 128 + mt * 16, 264);
  }
  __syncthreads();

  // ---- y = feat @ wd^T  [32 x 256], K=256; fused BN + residual + leaky ----
  for (int t = w; t < 32; t += 8) {
    int rt = t >> 4, ct = t & 15;
    v8f acc = {};
    const __bf16* ab = feat + rt * 16 * 264;
#pragma unroll
    for (int kc = 0; kc < 8; ++kc) {
      v16bf A = load_A(ab, lane, kc * 32, 264);
      v16bf Bf = load_Bg(Bwd, kc * 16 + ct, lane);
      WMMA_BF16(A, Bf, acc);
    }
    int c = ct * 16 + (lane & 15);
    float sc = gamma[c] * rsqrtf(var[c] + 1e-5f);
    float mn = mean[c], bt = beta[c];
    int p = p0 + rt * 16 + ((lane >> 4) << 3);
    const float* xr = x + ((size_t)b * CIC_C + c) * CIC_N + p;
    float* orow = out + ((size_t)b * CIC_C + c) * CIC_N + p;
#pragma unroll
    for (int rr = 0; rr < 8; ++rr) {
      float yv = (acc[rr] - mn) * sc + bt;
      float v = xr[rr] + yv;
      orow[rr] = v > 0.f ? v : 0.2f * v;
    }
  }
}

// ---------------- Host launcher -------------------------------------------
extern "C" void kernel_launch(void* const* d_in, const int* in_sizes, int n_in,
                              void* d_out, int out_size, void* d_ws, size_t ws_size,
                              hipStream_t stream) {
  (void)in_sizes; (void)n_in; (void)out_size; (void)ws_size;
  const float* x      = (const float*)d_in[0];
  const float* curves = (const float*)d_in[1];
  const float* w_att  = (const float*)d_in[2];
  const float* wa     = (const float*)d_in[3];
  const float* wb     = (const float*)d_in[4];
  const float* wc     = (const float*)d_in[5];
  const float* wn     = (const float*)d_in[6];
  const float* wl     = (const float*)d_in[7];
  const float* wd     = (const float*)d_in[8];
  const float* gamma  = (const float*)d_in[9];
  const float* beta   = (const float*)d_in[10];
  const float* mean   = (const float*)d_in[11];
  const float* var    = (const float*)d_in[12];
  float* out = (float*)d_out;

  char* ws = (char*)d_ws;
  size_t o = 0;
  auto take = [&](size_t bytes) -> char* {
    char* p = ws + o;
    o = (o + bytes + 1023) & ~(size_t)1023;
    return p;
  };
  float*  att  = (float*)take(131072 * 4);   // [B,n,l]
  float*  smlf = (float*)take(131072 * 4);   // softmax over l
  float*  smnf = (float*)take(131072 * 4);   // softmax over n
  float*  ci   = (float*)take(524288 * 4);   // [B,C,n]
  float*  ct   = (float*)take(262144 * 4);   // [B,C,l]
  float*  a_f  = (float*)take(262144 * 4);   // [B,mid,n]
  float*  bm_f = (float*)take(131072 * 4);   // [B,mid,l]
  float*  an_f = (float*)take(262144 * 4);   // [B,o,n]
  float*  bl_f = (float*)take(131072 * 4);   // [B,o,l]
  __bf16* Bwc  = (__bf16*)take(32768 * 2);   // K=256,N=128 packed
  __bf16* Bwd  = (__bf16*)take(65536 * 2);   // K=256,N=256 packed
  __bf16* Ba   = (__bf16*)take(262144 * 2);  // per-b K=128,N=128
  __bf16* Bb   = (__bf16*)take(131072 * 2);  // per-b K=128,N=64
  __bf16* Ban  = (__bf16*)take(262144 * 2);  // per-b K=128,N=128
  __bf16* Bbl  = (__bf16*)take(131072 * 2);  // per-b K=64, N=128

  // weight packing (batch-independent)
  cic_pack_b_kernel<<<8,  256, 0, stream>>>(wc, Bwc, 256, 128, 1, 256, 0, 64, 64);
  cic_pack_b_kernel<<<16, 256, 0, stream>>>(wd, Bwd, 256, 256, 1, 256, 0, 128, 128);

  // curve attention path
  cic_att_kernel<<<512, 256, 0, stream>>>(curves, w_att, att);
  cic_softmax_l_kernel<<<8, 256, 0, stream>>>(att, smlf, CIC_B * CIC_CN);
  cic_softmax_n_kernel<<<4, 256, 0, stream>>>(att, smnf, CIC_B * CIC_CL);
  cic_cict_kernel<<<CIC_B * CIC_C, 128, 0, stream>>>(curves, smlf, smnf, ci, ct);

  // small GEMMs: a = wa@ci, bmat = wb@ct, an = wn@a, bl = wl@bmat
  cic_mm_small_kernel<<<1024, 256, 0, stream>>>(wa, ci,   a_f,  128, 256, 128, 16 * 128 * 128);
  cic_mm_small_kernel<<<512,  256, 0, stream>>>(wb, ct,   bm_f, 128, 256, 64,  16 * 128 * 64);
  cic_mm_small_kernel<<<1024, 256, 0, stream>>>(wn, a_f,  an_f, 128, 128, 128, 16 * 128 * 128);
  cic_mm_small_kernel<<<512,  256, 0, stream>>>(wl, bm_f, bl_f, 128, 128, 64,  16 * 128 * 64);

  // pack per-batch B matrices for the fused kernel
  cic_pack_b_kernel<<<64, 256, 0, stream>>>(a_f,  Ba,  128, 128, 128, 1,  16384, 32, 512);
  cic_pack_b_kernel<<<32, 256, 0, stream>>>(bm_f, Bb,  128, 64,  64,  1,  8192,  16, 256);
  cic_pack_b_kernel<<<64, 256, 0, stream>>>(an_f, Ban, 128, 128, 1,   128, 16384, 32, 512);
  cic_pack_b_kernel<<<32, 256, 0, stream>>>(bl_f, Bbl, 64,  128, 1,   64, 8192,  16, 256);

  // fused per-point pipeline
  cic_main_kernel<<<dim3(CIC_N / CIC_PT, CIC_B), 256, 0, stream>>>(
      x, Bwc, Ba, Bb, Ban, Bbl, Bwd, gamma, beta, mean, var, out);
}
